// PlaceCellNetwork_63118839382369
// MI455X (gfx1250) — matrix operand order repaired and checked by reading, not compiled, not run on hardware
//
#include <hip/hip_runtime.h>
#include <hip/hip_fp16.h>
#include <math.h>

typedef __attribute__((ext_vector_type(16))) _Float16 v16h;
typedef __attribute__((ext_vector_type(8)))  _Float16 v8h;
typedef __attribute__((ext_vector_type(8)))  float    v8f;

#define B_      64
#define S_      512
#define N_      2048
#define GROUPS  4            // B / 16  (WMMA M = 16)
#define SLICES  32           // N / 64
#define COLS    64           // columns per workgroup
#define WAVES   4            // 4 waves * 16 cols = 64 cols
#define THREADS 128
#define NWG     (GROUPS * SLICES)   // 128 persistent workgroups
#define KTILES  (N_ / 32)           // 64 WMMA K-steps of 32
#define KHALF   (KTILES / 2)        // split-K: 2 independent accum chains

// ---------------------------------------------------------------------------
// Global sense-reversing barrier (persistent cooperative kernel).
// bar[0] = arrive count, bar[1] = generation. Release fence before arrival;
// after completion ALL threads execute an acq_rel agent fence so every
// wave/CU of this workgroup invalidates its L0 before reading peer data.
// ---------------------------------------------------------------------------
__device__ __forceinline__ void global_barrier(int* bar, int nwg) {
    __syncthreads();
    if (threadIdx.x == 0) {
        __threadfence();   // release all prior global writes (h, stats)
        int* cnt = bar;
        int* gen = bar + 1;
        int g = __hip_atomic_load(gen, __ATOMIC_RELAXED, __HIP_MEMORY_SCOPE_AGENT);
        int p = __hip_atomic_fetch_add(cnt, 1, __ATOMIC_ACQ_REL, __HIP_MEMORY_SCOPE_AGENT);
        if (p == nwg - 1) {
            __hip_atomic_store(cnt, 0, __ATOMIC_RELAXED, __HIP_MEMORY_SCOPE_AGENT);
            __hip_atomic_fetch_add(gen, 1, __ATOMIC_RELEASE, __HIP_MEMORY_SCOPE_AGENT);
        } else {
            while (__hip_atomic_load(gen, __ATOMIC_ACQUIRE, __HIP_MEMORY_SCOPE_AGENT) == g) {
                __builtin_amdgcn_s_sleep(2);
            }
        }
    }
    __syncthreads();
    __threadfence();       // acquire side for ALL threads (per-CU L0 invalidate)
}

// ---------------------------------------------------------------------------
// Pack W_rec (fp32, row-major [K=2048][N=2048]) into f16 B-fragments.
// Tile (nt, kt) covers cols nt*16..+15, K kt*32..+31.  Per-lane layout for
// V_WMMA_F32_16X16X32_F16 B (32x16):
//   lanes 0-15 : col = lane,    K = kt*32 + 0..7, 16..23
//   lanes 16-31: col = lane-16, K = kt*32 + 8..15, 24..31
// Stored lane-major: Wpack[((nt*KTILES + kt)*32 + lane)*16 + j]
// ---------------------------------------------------------------------------
__global__ void pack_w_kernel(const float* __restrict__ Wrec,
                              _Float16* __restrict__ Wpack) {
    int idx  = blockIdx.x * 256 + threadIdx.x;          // 0 .. N*N-1
    int j    = idx & 15;
    int lane = (idx >> 4) & 31;
    int tile = idx >> 9;
    int kt   = tile % KTILES;
    int nt   = tile / KTILES;
    int n    = nt * 16 + (lane & 15);
    int k    = kt * 32 + ((j >> 3) << 4) + ((lane >> 4) << 3) + (j & 7);
    Wpack[idx] = (_Float16)Wrec[(size_t)k * N_ + n];
}

// h0 (fp32 [B][N]) -> f16 H buffer; zero barrier + stats.
__global__ void init_state_kernel(const float* __restrict__ h0,
                                  _Float16* __restrict__ Hbuf,
                                  float* __restrict__ stats,
                                  int* __restrict__ bar) {
    int idx = blockIdx.x * 256 + threadIdx.x;
    if (idx < B_ * N_)                  Hbuf[idx]  = (_Float16)h0[idx];
    if (idx < GROUPS * SLICES * 16 * 2) stats[idx] = 0.0f;
    if (idx < 2)                        bar[idx]   = 0;
}

// ---------------------------------------------------------------------------
// Persistent scan kernel. Grid = 4 batch-groups x 32 N-slices = 128 WGs.
// Each WG: 4 waves, each wave owns a 16x16 WMMA output tile. K=2048 is split
// into two independent 1024-deep accumulator chains for XDL pipelining.
// ---------------------------------------------------------------------------
__launch_bounds__(THREADS)
__global__ void rnn_scan_kernel(const float* __restrict__ vel,    // [B,S,2]
                                const float* __restrict__ Wlin,   // [N,2]
                                const float* __restrict__ blin,   // [N]
                                const _Float16* __restrict__ Wpack,
                                _Float16* __restrict__ Hbuf,      // f16 [B][N]
                                float* __restrict__ stats,        // [G][SL][16] float2
                                int* __restrict__ bar,
                                float* __restrict__ out) {        // [B,S,N]
    __shared__ float zbuf[16][COLS];
    __shared__ float velS[16][2];
    __shared__ float rmaxS[16];
    __shared__ float pmS[16][8], psS[16][8];
    __shared__ float gmS[16], gsS[16];

    const int wg    = blockIdx.x;
    const int g     = wg / SLICES;
    const int slice = wg % SLICES;
    const int n0    = slice * COLS;
    const int th    = threadIdx.x;
    const int wv    = th >> 5;
    const int lane  = th & 31;
    const int ntile = slice * WAVES + wv;      // global 16-col tile, 0..127
    const int lhalf = (lane < 16) ? 0 : 1;

    // Per-lane constants for base-logit of this lane's output column.
    const int   nn  = n0 + wv * 16 + (lane & 15);
    const float wl0 = Wlin[nn * 2 + 0];
    const float wl1 = Wlin[nn * 2 + 1];
    const float bl  = blin[nn];

    const _Float16* Hg = Hbuf + (size_t)g * 16 * N_;        // this group's h rows
    // A-frag base: row M = lane&15; halves 0-7 at K+(lane<16?0:8), 8-15 at +16.
    const _Float16* ap = Hg + (size_t)(lane & 15) * N_ + lhalf * 8;
    // B-frag base for this wave's column tile.
    const _Float16* bp = Wpack + (size_t)ntile * KTILES * 512 + lane * 16;

    for (int t = 0; t < S_; ++t) {
        if (th < 32) {      // stage this step's velocities (16 rows x 2)
            int r = th >> 1, i = th & 1;
            velS[r][i] = vel[((size_t)(g * 16 + r) * S_ + t) * 2 + i];
        }
        __syncthreads();

        // ---- GEMM: 16x16 f32 tile; two independent split-K WMMA chains
        v8f c0 = {}, c1 = {};
        for (int kt = 0; kt < KHALF; ++kt) {
            union { v16h v; v8h h[2]; } a0, b0, a1, b1;
            a0.h[0] = *(const v8h*)(ap + kt * 32);
            a0.h[1] = *(const v8h*)(ap + kt * 32 + 16);
            b0.h[0] = *(const v8h*)(bp + (size_t)kt * 512);
            b0.h[1] = *(const v8h*)(bp + (size_t)kt * 512 + 8);
            a1.h[0] = *(const v8h*)(ap + (kt + KHALF) * 32);
            a1.h[1] = *(const v8h*)(ap + (kt + KHALF) * 32 + 16);
            b1.h[0] = *(const v8h*)(bp + (size_t)(kt + KHALF) * 512);
            b1.h[1] = *(const v8h*)(bp + (size_t)(kt + KHALF) * 512 + 8);
            c0 = __builtin_amdgcn_wmma_f32_16x16x32_f16(
                     false, a0.v, false, b0.v, (short)0, c0, false, false);
            c1 = __builtin_amdgcn_wmma_f32_16x16x32_f16(
                     false, a1.v, false, b1.v, (short)0, c1, false, false);
        }

        // ---- add base logits, spill z tile to LDS
        {
            int colrel = wv * 16 + (lane & 15);
            #pragma unroll
            for (int r = 0; r < 8; ++r) {
                int row = r + lhalf * 8;                    // C layout: M = r (+8)
                float z = c0[r] + c1[r]
                        + velS[row][0] * wl0 + velS[row][1] * wl1 + bl;
                zbuf[row][colrel] = z;
            }
        }
        __syncthreads();

        // ---- local softmax stats over this WG's 64 columns
        {
            int row = th & 15, part = th >> 4;
            float m = -3.0e38f;
            #pragma unroll
            for (int j2 = 0; j2 < 8; ++j2) m = fmaxf(m, zbuf[row][part * 8 + j2]);
            pmS[row][part] = m;
        }
        __syncthreads();
        if (th < 16) {
            float m = pmS[th][0];
            #pragma unroll
            for (int p = 1; p < 8; ++p) m = fmaxf(m, pmS[th][p]);
            rmaxS[th] = m;
        }
        __syncthreads();
        {
            int row = th & 15, part = th >> 4;
            float m = rmaxS[row], s = 0.0f;
            #pragma unroll
            for (int j2 = 0; j2 < 8; ++j2) s += __expf(zbuf[row][part * 8 + j2] - m);
            psS[row][part] = s;
        }
        __syncthreads();
        if (th < 16) {
            float s = 0.0f;
            #pragma unroll
            for (int p = 0; p < 8; ++p) s += psS[th][p];
            float2 st; st.x = rmaxS[th]; st.y = s;
            ((float2*)stats)[((size_t)g * SLICES + slice) * 16 + th] = st;
        }

        global_barrier(bar, NWG);   // all slices' stats published

        // ---- combine stats across 32 slices (log-sum-exp merge)
        {
            int row = th & 15, part = th >> 4;   // 8 parts x 4 slices
            float m = -3.0e38f, s = 0.0f;
            const float2* stg = (const float2*)stats + (size_t)g * SLICES * 16;
            #pragma unroll
            for (int q = 0; q < 4; ++q) {
                float2 st = stg[(size_t)(part * 4 + q) * 16 + row];
                float nm = fmaxf(m, st.x);
                s = s * __expf(m - nm) + st.y * __expf(st.x - nm);
                m = nm;
            }
            pmS[row][part] = m; psS[row][part] = s;
        }
        __syncthreads();
        if (th < 16) {
            float m = -3.0e38f, s = 0.0f;
            #pragma unroll
            for (int p = 0; p < 8; ++p) {
                float nm = fmaxf(m, pmS[th][p]);
                s = s * __expf(m - nm) + psS[th][p] * __expf(pmS[th][p] - nm);
                m = nm;
            }
            gmS[th] = m; gsS[th] = 1.0f / s;
        }
        __syncthreads();

        // ---- normalize: write fp32 output + f16 h for next step
        {
            int row = th >> 3;
            int cb  = (th & 7) * 8;
            float m = gmS[row], is = gsS[row];
            int b = g * 16 + row;
            float*    op = out  + ((size_t)b * S_ + t) * N_ + n0 + cb;
            _Float16* hp = Hbuf +  (size_t)b * N_        + n0 + cb;
            #pragma unroll
            for (int j2 = 0; j2 < 8; ++j2) {
                float h = __expf(zbuf[row][cb + j2] - m) * is;
                op[j2] = h;
                hp[j2] = (_Float16)h;
            }
        }

        global_barrier(bar, NWG);   // h(t+1) visible before next GEMM
    }
}

// ---------------------------------------------------------------------------
extern "C" void kernel_launch(void* const* d_in, const int* in_sizes, int n_in,
                              void* d_out, int out_size, void* d_ws, size_t ws_size,
                              hipStream_t stream) {
    (void)in_sizes; (void)n_in; (void)out_size; (void)ws_size;
    const float* vel  = (const float*)d_in[0];   // [64,512,2]
    const float* h0   = (const float*)d_in[1];   // [64,2048]
    const float* Wlin = (const float*)d_in[2];   // [2048,2]
    const float* blin = (const float*)d_in[3];   // [2048]
    const float* Wrec = (const float*)d_in[4];   // [2048,2048]
    float* out = (float*)d_out;                  // [64,512,2048]

    char* ws = (char*)d_ws;
    _Float16* Wpack = (_Float16*)ws;                                   // 8 MB
    _Float16* Hbuf  = (_Float16*)(ws + (size_t)N_ * N_ * 2);           // 256 KB
    float*    stats = (float*)(ws + (size_t)N_ * N_ * 2 + (size_t)B_ * N_ * 2);
    int*      bar   = (int*)((char*)stats + (size_t)GROUPS * SLICES * 16 * 2 * sizeof(float));

    pack_w_kernel<<<(N_ * N_) / 256, 256, 0, stream>>>(Wrec, Wpack);
    init_state_kernel<<<(B_ * N_ + 255) / 256, 256, 0, stream>>>(h0, Hbuf, stats, bar);
    rnn_scan_kernel<<<NWG, THREADS, 0, stream>>>(vel, Wlin, blin, Wpack,
                                                 Hbuf, stats, bar, out);
}
